// AdaptiveGraphConstructor_78142634983588
// MI455X (gfx1250) — compile-verified, matrix operand order checked by reading, not run on hardware
//
#include <hip/hip_runtime.h>
#include <hip/hip_bf16.h>

// ---------------------------------------------------------------------------
// AdaptiveGraphConstructor for MI455X (gfx1250, wave32, WMMA)
//
// Stage 1: enc = relu(X@W1^T+b1)@W2^T+b2                 (tiny, VALU+LDS)
// Stage 2: Ap,Bp,Af,Bf = enc @ {Wp1,Wf1} halves          (tiny, VALU+LDS)
// Stage 3: per 16x16 (i,j) tile (one wave each):
//            feat = relu(Af[i]+Bf[j]+bf1) @ Wf2^T + bf2  via v_wmma_f32_16x16x32_f16
//            logit = relu(Ap[i]+Bp[j]+bp1) . wp2 + bp2   via VALU + shfl reduce
//          D fragments staged through LDS -> coalesced global_store_b128.
//          ~330 MB of output writes dominate (14us @ 23.3 TB/s) -> streaming.
// ---------------------------------------------------------------------------

typedef __attribute__((ext_vector_type(16))) _Float16 v16h;
typedef __attribute__((ext_vector_type(8)))  float    v8f;

#define NNODES   1536
#define NODE_DIM 128
#define HIDDEN   64
#define EDGE_DIM 32
#define NT       96                 // NNODES/16
#define NTILES   4656               // NT*(NT+1)/2

static constexpr long long PAIRS   = (long long)NNODES * (NNODES - 1) / 2; // 1178880
static constexpr long long IDX1OFF = 2 * PAIRS;   // row1 of edge_index
static constexpr long long MASKOFF = 4 * PAIRS;   // edge_mask
static constexpr long long ATTROFF = 6 * PAIRS;   // edge_attr [2P, 32]

// ---------------- Stage 1: node encoder ----------------
__global__ __launch_bounds__(64)
void enc_kernel(const float* __restrict__ X,
                const float* __restrict__ W1, const float* __restrict__ b1,
                const float* __restrict__ W2, const float* __restrict__ b2,
                float* __restrict__ enc) {
  __shared__ float xs[NODE_DIM];
  __shared__ float h1[HIDDEN];
  const int row = blockIdx.x, t = threadIdx.x;
  xs[t]      = X[(size_t)row * NODE_DIM + t];
  xs[t + 64] = X[(size_t)row * NODE_DIM + 64 + t];
  __syncthreads();
  float a = b1[t];
  const float* w1r = W1 + (size_t)t * NODE_DIM;
#pragma unroll 8
  for (int k = 0; k < NODE_DIM; ++k) a = fmaf(w1r[k], xs[k], a);
  h1[t] = fmaxf(a, 0.f);
  __syncthreads();
  float e = b2[t];
  const float* w2r = W2 + (size_t)t * HIDDEN;
#pragma unroll 8
  for (int k = 0; k < HIDDEN; ++k) e = fmaf(w2r[k], h1[k], e);
  enc[(size_t)row * HIDDEN + t] = e;
}

// ---------------- Stage 2: pair projections (biases folded in later) -------
__global__ __launch_bounds__(64)
void proj_kernel(const float* __restrict__ enc,
                 const float* __restrict__ Wp1, const float* __restrict__ Wf1,
                 float* __restrict__ Ap, float* __restrict__ Bp,
                 float* __restrict__ Af, float* __restrict__ Bf) {
  __shared__ float es[HIDDEN];
  const int row = blockIdx.x, t = threadIdx.x;
  es[t] = enc[(size_t)row * HIDDEN + t];
  __syncthreads();
  float ap = 0.f, bp = 0.f, af = 0.f, bf = 0.f;
  const float* wpr = Wp1 + (size_t)t * 2 * HIDDEN;
  const float* wfr = Wf1 + (size_t)t * 2 * HIDDEN;
#pragma unroll 8
  for (int k = 0; k < HIDDEN; ++k) {
    const float e = es[k];
    ap = fmaf(wpr[k], e, ap);
    bp = fmaf(wpr[HIDDEN + k], e, bp);
    af = fmaf(wfr[k], e, af);
    bf = fmaf(wfr[HIDDEN + k], e, bf);
  }
  Ap[(size_t)row * HIDDEN + t] = ap;
  Bp[(size_t)row * HIDDEN + t] = bp;
  Af[(size_t)row * HIDDEN + t] = af;
  Bf[(size_t)row * HIDDEN + t] = bf;
}

// ---------------- Stage 3: pair tiles via WMMA ----------------
__device__ __forceinline__ v8f wmma_f16(v16h a, v16h b, v8f c) {
  return __builtin_amdgcn_wmma_f32_16x16x32_f16(
      /*neg_a=*/false, a, /*neg_b=*/false, b,
      /*c_mod=*/(short)0, c, /*reuse_a=*/false, /*reuse_b=*/false);
}

__global__ __launch_bounds__(32)
void pair_kernel(const float* __restrict__ Af, const float* __restrict__ Bf,
                 const float* __restrict__ Ap, const float* __restrict__ Bp,
                 const float* __restrict__ bp1, const float* __restrict__ wp2,
                 const float* __restrict__ bp2, const float* __restrict__ bf1,
                 const float* __restrict__ Wf2, const float* __restrict__ bf2,
                 float* __restrict__ out) {
  __shared__ float sattr[16 * EDGE_DIM];  // 16 pairs x 32 feat floats (2 KB)

  const int lane = threadIdx.x;           // 0..31 (wave32)
  const int T = blockIdx.x;               // triangular tile id, bi <= bj
  // decode (bi, bj) from triangular index: f(b) = b*NT - b*(b-1)/2
  int bi;
  {
    const float tf = (float)T;
    bi = (int)((193.f - sqrtf(193.f * 193.f - 8.f * tf)) * 0.5f);
    if (bi < 0) bi = 0;
    if (bi > NT - 1) bi = NT - 1;
    while (bi > 0 && (long long)bi * NT - (long long)bi * (bi - 1) / 2 > T) --bi;
    while ((long long)(bi + 1) * NT - (long long)(bi + 1) * bi / 2 <= T) ++bi;
  }
  const long long rowbase = (long long)bi * NT - (long long)bi * (bi - 1) / 2;
  const int bj = bi + (int)(T - rowbase);

  const int m  = lane & 15;   // A row-in-frag (= pair col c), D col n, B col n
  const int hi = lane >> 4;   // K half selector per ISA A/B VGPR layout
  const int jm = bj * 16 + m; // the j-node this lane owns for frag building

  // per-half-lane K indices for 16x32 f16 A/B fragments:
  // halves h<8  -> K = h   + 8*hi
  // halves h>=8 -> K = h+8 + 8*hi   (covers 0..31 across the wave)
  int kidx[16];
#pragma unroll
  for (int h = 0; h < 16; ++h) kidx[h] = ((h < 8) ? h : h + 8) + hi * 8;

  // B fragments of Wf2^T [64x32], constant across tiles/rows: [kc][nc]
  v16h bW[2][2];
#pragma unroll
  for (int h = 0; h < 16; ++h) {
    const int k0 = kidx[h], k1 = kidx[h] + 32;
    bW[0][0][h] = (_Float16)Wf2[(size_t)m * HIDDEN + k0];
    bW[0][1][h] = (_Float16)Wf2[(size_t)(16 + m) * HIDDEN + k0];
    bW[1][0][h] = (_Float16)Wf2[(size_t)m * HIDDEN + k1];
    bW[1][1][h] = (_Float16)Wf2[(size_t)(16 + m) * HIDDEN + k1];
  }

  // hoisted per-lane partials: Bf[j]+bf1 (A-frag) and Bp[j]+bp1, wp2 (logit)
  float bsum[2][16];
#pragma unroll
  for (int h = 0; h < 16; ++h) {
    const int k0 = kidx[h];
    bsum[0][h] = Bf[(size_t)jm * HIDDEN + k0] + bf1[k0];
    bsum[1][h] = Bf[(size_t)jm * HIDDEN + k0 + 32] + bf1[k0 + 32];
  }
  const int kb = hi * 32;
  float bpv[32], wpv[32];
#pragma unroll 8
  for (int k = 0; k < 32; ++k) {
    bpv[k] = Bp[(size_t)jm * HIDDEN + kb + k] + bp1[kb + k];
    wpv[k] = wp2[kb + k];
  }
  const float bp2v  = bp2[0];
  const float c0add = bf2[m];
  const float c1add = bf2[16 + m];

  for (int r = 0; r < 16; ++r) {
    const int i = bi * 16 + r;
    const float* Afi = Af + (size_t)i * HIDDEN;

    // prefetch next row-block's broadcast rows (global_prefetch_b8)
    __builtin_prefetch(Af + (size_t)(i + 1) * HIDDEN, 0, 0);
    __builtin_prefetch(Ap + (size_t)(i + 1) * HIDDEN, 0, 0);

    // A fragments: rows = 16 pairs (i fixed, j = bj*16 + 0..15)
    v16h a0, a1;
#pragma unroll
    for (int h = 0; h < 16; ++h) {
      a0[h] = (_Float16)fmaxf(Afi[kidx[h]]      + bsum[0][h], 0.f);
      a1[h] = (_Float16)fmaxf(Afi[kidx[h] + 32] + bsum[1][h], 0.f);
    }
    v8f acc0 = {}, acc1 = {};
    acc0 = wmma_f16(a0, bW[0][0], acc0);
    acc0 = wmma_f16(a1, bW[1][0], acc0);
    acc1 = wmma_f16(a0, bW[0][1], acc1);
    acc1 = wmma_f16(a1, bW[1][1], acc1);

    // logits: lane (m,hi) covers pair col m, K half hi; reduce across halves
    const float* Api = Ap + (size_t)i * HIDDEN + kb;
    float lac = 0.f;
#pragma unroll 8
    for (int k = 0; k < 32; ++k)
      lac = fmaf(fmaxf(Api[k] + bpv[k], 0.f), wpv[k], lac);
    lac += __shfl_xor(lac, 16, 32);
    const float logit = lac + bp2v;
    const unsigned mbits =
        (unsigned)__ballot(logit > 0.f) & 0xFFFFu;   // sigmoid>0.5 <=> logit>0

    const long long pbase = (long long)i * NNODES -
                            (long long)i * (i + 1) / 2 - i - 1; // p = pbase + j

    // edge_index + edge_mask (lanes 0..15, one pair column each)
    if (hi == 0) {
      const int j = jm;
      if (j > i) {
        const long long p = pbase + j;
        const float fi = (float)i, fj = (float)j;
        const float fm = ((mbits >> m) & 1u) ? 1.f : 0.f;
        out[2 * p]               = fi;
        out[2 * p + 1]           = fj;
        out[IDX1OFF + 2 * p]     = fj;
        out[IDX1OFF + 2 * p + 1] = fi;
        out[MASKOFF + 2 * p]     = fm;
        out[MASKOFF + 2 * p + 1] = fm;
      }
    }

    // ---- edge_attr: stage masked+biased D frags into LDS ----
    // D layout: lane -> (n = m, M = v + 8*hi); LDS is pair-major [c][n]
#pragma unroll
    for (int v = 0; v < 8; ++v) {
      const int c = v + hi * 8;
      const bool mk = (mbits >> c) & 1u;
      sattr[c * EDGE_DIM + m]      = mk ? (acc0[v] + c0add) : 0.f;
      sattr[c * EDGE_DIM + 16 + m] = mk ? (acc1[v] + c1add) : 0.f;
    }
    __syncthreads();  // single-wave WG: barrier -> NOP, keeps ds-cnt ordering

    // ---- stream out: 2 duplicated rows of 32 floats per pair, b128 stores --
    // 16 pairs x 16 float4 slots = 256 stores; 8 per lane, fully coalesced.
#pragma unroll
    for (int it = 0; it < 8; ++it) {
      const int e = it * 32 + lane;   // 0..255
      const int c = e >> 4;           // pair column 0..15
      const int f = e & 15;           // float4 slot (two copies of 8)
      const int j = bj * 16 + c;
      if (j > i) {
        const long long p = pbase + j;
        const size_t ab = (size_t)ATTROFF + (size_t)(2 * p) * EDGE_DIM +
                          ((f & 8) ? EDGE_DIM : 0) + (size_t)(f & 7) * 4;
        *(float4*)(out + ab) =
            *(const float4*)(sattr + c * EDGE_DIM + (f & 7) * 4);
      }
    }
    __syncthreads();  // protect sattr before next row-block overwrites it
  }
}

// ---------------------------------------------------------------------------
extern "C" void kernel_launch(void* const* d_in, const int* in_sizes, int n_in,
                              void* d_out, int out_size, void* d_ws, size_t ws_size,
                              hipStream_t stream) {
  const float* X   = (const float*)d_in[0];
  const float* W1  = (const float*)d_in[1];
  const float* b1  = (const float*)d_in[2];
  const float* W2  = (const float*)d_in[3];
  const float* b2  = (const float*)d_in[4];
  const float* Wp1 = (const float*)d_in[5];
  const float* bp1 = (const float*)d_in[6];
  const float* wp2 = (const float*)d_in[7];
  const float* bp2 = (const float*)d_in[8];
  const float* Wf1 = (const float*)d_in[9];
  const float* bf1 = (const float*)d_in[10];
  const float* Wf2 = (const float*)d_in[11];
  const float* bf2 = (const float*)d_in[12];
  float* out = (float*)d_out;

  float* enc = (float*)d_ws;                       // [1536*64]
  float* Ap  = enc + (size_t)NNODES * HIDDEN;
  float* Bp  = Ap  + (size_t)NNODES * HIDDEN;
  float* Af  = Bp  + (size_t)NNODES * HIDDEN;
  float* Bf  = Af  + (size_t)NNODES * HIDDEN;      // total ~2 MB scratch

  enc_kernel<<<NNODES, 64, 0, stream>>>(X, W1, b1, W2, b2, enc);
  proj_kernel<<<NNODES, 64, 0, stream>>>(enc, Wp1, Wf1, Ap, Bp, Af, Bf);
  pair_kernel<<<NTILES, 32, 0, stream>>>(Af, Bf, Ap, Bp, bp1, wp2, bp2,
                                         bf1, Wf2, bf2, out);
}
